// Delta_63926293233951
// MI455X (gfx1250) — matrix operand order repaired and checked by reading, not compiled, not run on hardware
//
#include <hip/hip_runtime.h>

// ---------------------------------------------------------------------------
// DeltaNet-style pipeline for MI455X (gfx1250), wave32 + WMMA.
//   g = x@Wg^T + bg  -> q,k,v   (bf16 WMMA, double-buffered async global->LDS)
//   lr = x@Wlr^T+blr ; k <- k/||k||
//   chunked delta recurrence (fp32 WMMA 16x16x4) -> kv = sum_t k_t (lr u_t)^T
//   y1 = q @ kv  (bf16 WMMA, batched)
//   out = y1 @ Wo^T + bo (bf16 WMMA)
// ---------------------------------------------------------------------------

typedef __attribute__((ext_vector_type(16))) __bf16 v16bf;
typedef __attribute__((ext_vector_type(8)))  float  v8f;
typedef __attribute__((ext_vector_type(2)))  float  v2f;

__device__ __forceinline__ unsigned short f2bf(float f) {
  union { float f; unsigned int u; } x; x.f = f;
  unsigned int r = x.u + 0x7FFFu + ((x.u >> 16) & 1u);   // round-to-nearest-even
  return (unsigned short)(r >> 16);
}

// -------------------- f32 -> bf16 (strided, cols==512) ---------------------
__global__ void cvt_f32_bf16(const float* __restrict__ src, unsigned short* __restrict__ dst,
                             long total, long lds, long ldd) {
  for (long i = (long)blockIdx.x * blockDim.x + threadIdx.x; i < total;
       i += (long)gridDim.x * blockDim.x) {
    long r = i >> 9, c = i & 511;
    dst[r * ldd + c] = f2bf(src[r * lds + c]);
  }
}

// -------------------- bf16 GEMM: C = A @ op(B) + bias ----------------------
// TB=true : B is [Nn,K] row-major, compute A@B^T (fully async-staged)
// TB=false: B is [K,Nn]  row-major, compute A@B (A async, B transposed in LDS)
// Block = 128 threads (4 waves) computing a 64x64 tile. LDS tiles are
// double-buffered: tile k0+32 streams in via global_load_async_to_lds_b128
// (ASYNCcnt) while the 4 WMMAs of tile k0 execute; s_wait_asynccnt leaves the
// next tile's copies in flight. All dims are multiples of 64 in every call.
template <bool TB>
__global__ void gemm_bf16(const unsigned short* __restrict__ A,
                          const unsigned short* __restrict__ B,
                          const float* __restrict__ bias,
                          float* __restrict__ Cm,
                          int M, int Nn, int K, int lda, int ldb, int ldc,
                          long strA, long strB, long strC) {
  A  += (long)blockIdx.z * strA;
  B  += (long)blockIdx.z * strB;
  Cm += (long)blockIdx.z * strC;

  __shared__ __align__(16) unsigned short sA[2][64][32];  // [buf][m][k]
  __shared__ __align__(16) unsigned short sB[2][64][32];  // [buf][n][k]

  const int tid = threadIdx.x;
  const int lane = tid & 31;
  const int wave = tid >> 5;                 // 0..3
  const int bm0 = blockIdx.y * 64;
  const int bn0 = blockIdx.x * 64;
  const int wm = (wave >> 1) * 32;
  const int wn = (wave & 1) * 32;
  const int l15 = lane & 15, lh = lane >> 4;

  // issue async copies of one 64x32 tile (256 x 16B chunks, 2 per thread)
  auto issueA = [&](int buf, int kk) {
    for (int c = tid; c < 256; c += 128) {
      int row = c >> 2, ko = (c & 3) * 8;
      const unsigned short* gp = A + (long)(bm0 + row) * lda + kk + ko;
      unsigned lo = (unsigned)(size_t)&sA[buf][row][ko];  // low 32b = LDS addr
      asm volatile("global_load_async_to_lds_b128 %0, %1, off"
                   :: "v"(lo), "v"(gp) : "memory");
      if (kk + 64 < K) __builtin_prefetch(gp + 64, 0, 0); // global_prefetch_b8
    }
  };
  auto issueB = [&](int buf, int kk) {
    for (int c = tid; c < 256; c += 128) {
      int row = c >> 2, ko = (c & 3) * 8;
      const unsigned short* gp = B + (long)(bn0 + row) * ldb + kk + ko;
      unsigned lo = (unsigned)(size_t)&sB[buf][row][ko];
      asm volatile("global_load_async_to_lds_b128 %0, %1, off"
                   :: "v"(lo), "v"(gp) : "memory");
      if (kk + 64 < K) __builtin_prefetch(gp + 64, 0, 0);
    }
  };

  // prologue: tile 0 into buffer 0
  issueA(0, 0);
  if (TB) issueB(0, 0);

  v8f acc[2][2] = {};
  for (int k0 = 0; k0 < K; k0 += 32) {
    const int cur = (k0 >> 5) & 1, nxt = cur ^ 1;
    const bool has_next = (k0 + 32 < K);
    if (has_next) {                       // stream next tile into other buffer
      issueA(nxt, k0 + 32);
      if (TB) issueB(nxt, k0 + 32);
    }
    if (!TB) {                            // stage B[k][n] transposed -> sB[n][k]
      for (int c = tid; c < 256; c += 128) {
        int kr = c >> 3, no = (c & 7) * 8;
        union { uint4 v; unsigned short h[8]; } u;
        u.v = *(const uint4*)(B + (long)(k0 + kr) * ldb + bn0 + no);
#pragma unroll
        for (int e = 0; e < 8; ++e) sB[cur][no + e][kr] = u.h[e];
      }
    }
    // wait for current tile only; next tile's copies stay in flight
    if (has_next) {
      if (TB) asm volatile("s_wait_asynccnt 0x4" ::: "memory");
      else    asm volatile("s_wait_asynccnt 0x2" ::: "memory");
    } else {
      asm volatile("s_wait_asynccnt 0x0" ::: "memory");
    }
    __syncthreads();

    // ---- fragments from LDS (each half-fragment = 16 contiguous bytes) ----
    union Frag { unsigned int u[8]; uint4 q[2]; v16bf v; } af[2], bfr[2];
#pragma unroll
    for (int i = 0; i < 2; ++i) {
      int m = wm + i * 16 + l15;
      af[i].q[0] = *(const uint4*)&sA[cur][m][lh * 8];        // k = lh*8+0..7
      af[i].q[1] = *(const uint4*)&sA[cur][m][16 + lh * 8];   // +16
      int n = wn + i * 16 + l15;
      bfr[i].q[0] = *(const uint4*)&sB[cur][n][lh * 8];       // k = lh*8+0..7
      bfr[i].q[1] = *(const uint4*)&sB[cur][n][16 + lh * 8];  // +16
    }
#pragma unroll
    for (int i = 0; i < 2; ++i)
#pragma unroll
      for (int jj = 0; jj < 2; ++jj)
        acc[i][jj] = __builtin_amdgcn_wmma_f32_16x16x32_bf16(
            false, af[i].v, false, bfr[jj].v, (short)0, acc[i][jj], false, false);
    __syncthreads();   // all reads of `cur` done before it is refilled
  }
  // ---- store: D row m = half*8 + j, col n = l15 ----
#pragma unroll
  for (int i = 0; i < 2; ++i)
#pragma unroll
    for (int jj = 0; jj < 2; ++jj) {
      int col = bn0 + wn + jj * 16 + l15;
      float bv = bias ? bias[col] : 0.0f;
#pragma unroll
      for (int j = 0; j < 8; ++j) {
        int row = bm0 + wm + i * 16 + lh * 8 + j;
        Cm[(long)row * ldc + col] = acc[i][jj][j] + bv;
      }
    }
}

// -------------------- lr = x@Wlr^T + blr ; k <- k / ||k|| ------------------
__global__ void lr_norm_kernel(const float* __restrict__ x, const float* __restrict__ Wlr,
                               const float* __restrict__ blr, float* __restrict__ g,
                               float* __restrict__ lr, int C) {
  const int row = blockIdx.x;
  const float* xr = x + (long)row * C;
  float* kr = g + (long)row * 1536 + 512;
  __shared__ float red[2][8];
  float dp = 0.f, ss = 0.f;
  for (int c = threadIdx.x; c < C; c += blockDim.x) {
    dp += xr[c] * Wlr[c];
    float kv = kr[c];
    ss += kv * kv;
  }
  for (int off = 16; off; off >>= 1) {
    dp += __shfl_down(dp, off);
    ss += __shfl_down(ss, off);
  }
  int wv = threadIdx.x >> 5, ln = threadIdx.x & 31;
  if (ln == 0) { red[0][wv] = dp; red[1][wv] = ss; }
  __syncthreads();
  if (threadIdx.x == 0) {
    float d = 0.f, s = 0.f;
    for (int w = 0; w < 8; ++w) { d += red[0][w]; s += red[1][w]; }
    lr[row] = d + blr[0];
    red[1][0] = rsqrtf(fmaxf(s, 1e-24f));
  }
  __syncthreads();
  float inv = red[1][0];
  for (int c = threadIdx.x; c < C; c += blockDim.x) kr[c] *= inv;
}

// -------------------- chunked delta-rule recurrence (fp32 WMMA) ------------
// grid = (C/64 d-slices, N batches), block = 256 threads (8 waves).
// Per chunk (L=64): W = K_c @ S_slice; A = tril(K_c K_c^T,-1);
// forward-substitute U; S += K_c^T U; kv += K_c^T (lr .* U).
__global__ void recurrence_kernel(const float* __restrict__ g, const float* __restrict__ lr,
                                  float* __restrict__ Sws, float* __restrict__ kv) {
  const int n = blockIdx.y;
  const int ds = blockIdx.x * 64;
  const float* Kp = g + (long)n * 2048 * 1536 + 512;   // k cols, row stride 1536
  const float* Vp = g + (long)n * 2048 * 1536 + 1024;  // v cols
  const float* lrp = lr + (long)n * 2048;
  float* S = Sws + (long)(n * gridDim.x + blockIdx.x) * (512 * 64);  // [512][64]
  float* kvp = kv + (long)n * 512 * 512;                             // [512][512]

  __shared__ float As[64 * 64];
  __shared__ float Ws[64 * 64];
  __shared__ float Us[64 * 64];

  const int lane = threadIdx.x & 31;
  const int wave = threadIdx.x >> 5;
  const int l15 = lane & 15, lh = lane >> 4;

  // zero state + kv slice (graph-replay safe: re-init every launch)
  for (int i = threadIdx.x; i < 512 * 64; i += blockDim.x) S[i] = 0.0f;
  for (int i = threadIdx.x; i < 512 * 64; i += blockDim.x)
    kvp[(long)(i >> 6) * 512 + ds + (i & 63)] = 0.0f;
  __syncthreads();

  for (int ch = 0; ch < 2048 / 64; ++ch) {
    const int t0 = ch * 64;

    // ---- Phase 1: A (jobs 0..15) and W (jobs 16..31), 16x16 tiles ----
    for (int job = wave; job < 32; job += 8) {
      const int tile = job & 15;
      const int tm = (tile >> 2) * 16;  // t offset in chunk
      const int tn = (tile & 3) * 16;   // s offset (A) / d offset (W)
      v8f acc = {};
      if (job < 16) {
        if (tn <= tm) {  // strictly-lower block triangle only
          const float* ar = Kp + (long)(t0 + tm + l15) * 1536;
          const float* br = Kp + (long)(t0 + tn + l15) * 1536;
          for (int k0 = 0; k0 < 512; k0 += 4) {
            v2f a, b;
            a[0] = ar[k0 + lh * 2 + 0]; a[1] = ar[k0 + lh * 2 + 1];
            b[0] = br[k0 + lh * 2 + 0]; b[1] = br[k0 + lh * 2 + 1];
            acc = __builtin_amdgcn_wmma_f32_16x16x4_f32(false, a, false, b,
                                                        (short)0, acc, false, false);
          }
        }
#pragma unroll
        for (int j = 0; j < 8; ++j) {
          int tt = tm + lh * 8 + j, ss = tn + l15;
          As[tt * 64 + ss] = (ss < tt) ? acc[j] : 0.0f;
        }
      } else {
        const float* ar = Kp + (long)(t0 + tm + l15) * 1536;
        for (int k0 = 0; k0 < 512; k0 += 4) {
          v2f a, b;
          a[0] = ar[k0 + lh * 2 + 0]; a[1] = ar[k0 + lh * 2 + 1];
          b[0] = S[(k0 + lh * 2 + 0) * 64 + tn + l15];
          b[1] = S[(k0 + lh * 2 + 1) * 64 + tn + l15];
          acc = __builtin_amdgcn_wmma_f32_16x16x4_f32(false, a, false, b,
                                                      (short)0, acc, false, false);
        }
#pragma unroll
        for (int j = 0; j < 8; ++j)
          Ws[(tm + lh * 8 + j) * 64 + tn + l15] = acc[j];
      }
    }
    __syncthreads();

    // ---- Phase 2: forward substitution u_t = v_t - W_t - sum_{s<t} A[t,s] u_s
    for (int t = 0; t < 64; ++t) {
      if (threadIdx.x < 64) {
        int d = threadIdx.x;
        float val = Vp[(long)(t0 + t) * 1536 + ds + d] - Ws[t * 64 + d];
        for (int s = 0; s < t; ++s) val -= As[t * 64 + s] * Us[s * 64 + d];
        Us[t * 64 + d] = val;
      }
      __syncthreads();
    }

    // ---- Phase 3: S += K^T U ; kv += K^T (lr .* U) ----  (128 tiles)
    for (int tile = wave; tile < 128; tile += 8) {
      const int cm = (tile >> 2) * 16;  // c offset 0..496
      const int dn = (tile & 3) * 16;   // d offset 0..48
      v8f accS, accK;
#pragma unroll
      for (int j = 0; j < 8; ++j) {
        int row = cm + lh * 8 + j;
        accS[j] = S[row * 64 + dn + l15];
        accK[j] = kvp[(long)row * 512 + ds + dn + l15];
      }
      for (int k0 = 0; k0 < 64; k0 += 4) {
        int t_a = t0 + k0 + lh * 2;
        v2f a, bU, bP;
        a[0] = Kp[(long)(t_a + 0) * 1536 + cm + l15];
        a[1] = Kp[(long)(t_a + 1) * 1536 + cm + l15];
        bU[0] = Us[(k0 + lh * 2 + 0) * 64 + dn + l15];
        bU[1] = Us[(k0 + lh * 2 + 1) * 64 + dn + l15];
        bP[0] = bU[0] * lrp[t_a + 0];
        bP[1] = bU[1] * lrp[t_a + 1];
        accS = __builtin_amdgcn_wmma_f32_16x16x4_f32(false, a, false, bU,
                                                     (short)0, accS, false, false);
        accK = __builtin_amdgcn_wmma_f32_16x16x4_f32(false, a, false, bP,
                                                     (short)0, accK, false, false);
      }
#pragma unroll
      for (int j = 0; j < 8; ++j) {
        int row = cm + lh * 8 + j;
        S[row * 64 + dn + l15] = accS[j];
        kvp[(long)row * 512 + ds + dn + l15] = accK[j];
      }
    }
    __syncthreads();
  }
}

// ---------------------------------------------------------------------------
extern "C" void kernel_launch(void* const* d_in, const int* in_sizes, int n_in,
                              void* d_out, int out_size, void* d_ws, size_t ws_size,
                              hipStream_t stream) {
  (void)in_sizes; (void)n_in; (void)out_size; (void)ws_size;
  const float* x   = (const float*)d_in[0];
  const float* Wg  = (const float*)d_in[1];
  const float* bg  = (const float*)d_in[2];
  const float* Wlr = (const float*)d_in[3];
  const float* blr = (const float*)d_in[4];
  const float* Wo  = (const float*)d_in[5];
  const float* bo  = (const float*)d_in[6];
  float* out = (float*)d_out;

  const long N = 8, T = 2048, C = 512, M = N * T, C3 = 3 * C;

  char* p = (char*)d_ws;
  auto take = [&](size_t bytes) {
    char* r = p;
    p += (bytes + 255) & ~(size_t)255;
    return r;
  };
  float* g_f  = (float*)take((size_t)M * C3 * 4);        // q|k|v  [M,1536]
  float* lr_f = (float*)take((size_t)M * 4);             // lr     [M]
  float* S_f  = (float*)take((size_t)64 * 512 * 64 * 4); // per-block state slices
  float* kv_f = (float*)take((size_t)N * C * C * 4);     // kv     [N,C,C]
  float* y1_f = (float*)take((size_t)M * C * 4);         // y1     [M,C]
  unsigned short* xb  = (unsigned short*)take((size_t)M * C * 2);
  unsigned short* wgb = (unsigned short*)take((size_t)C3 * C * 2);
  unsigned short* wob = (unsigned short*)take((size_t)C * C * 2);
  unsigned short* qb  = (unsigned short*)take((size_t)M * C * 2);
  unsigned short* kvb = (unsigned short*)take((size_t)N * C * C * 2);
  unsigned short* y1b = xb;  // xb dead after the g-projection; reuse

  dim3 b256(256);
  cvt_f32_bf16<<<dim3((unsigned)((M * C + 1023) / 1024)), b256, 0, stream>>>(x, xb, M * C, C, C);
  cvt_f32_bf16<<<dim3((unsigned)((C3 * C + 1023) / 1024)), b256, 0, stream>>>(Wg, wgb, C3 * C, C, C);
  cvt_f32_bf16<<<dim3((unsigned)((C * C + 1023) / 1024)), b256, 0, stream>>>(Wo, wob, C * C, C, C);

  // g = x @ Wg^T + bg
  gemm_bf16<true><<<dim3((unsigned)(C3 / 64), (unsigned)(M / 64), 1), dim3(128), 0, stream>>>(
      xb, wgb, bg, g_f, (int)M, (int)C3, (int)C, (int)C, (int)C, (int)C3, 0, 0, 0);

  lr_norm_kernel<<<dim3((unsigned)M), b256, 0, stream>>>(x, Wlr, blr, g_f, lr_f, (int)C);

  // qb = bf16(q) (strided out of g)
  cvt_f32_bf16<<<dim3((unsigned)((M * C + 1023) / 1024)), b256, 0, stream>>>(g_f, qb, M * C, C3, C);

  // chunked delta recurrence -> kv
  recurrence_kernel<<<dim3(8, 8), b256, 0, stream>>>(g_f, lr_f, S_f, kv_f);

  cvt_f32_bf16<<<dim3((unsigned)((N * C * C + 1023) / 1024)), b256, 0, stream>>>(kv_f, kvb, N * C * C, C, C);

  // y1 = q @ kv (batched over N)
  gemm_bf16<false><<<dim3((unsigned)(C / 64), (unsigned)(T / 64), (unsigned)N), dim3(128), 0, stream>>>(
      qb, kvb, nullptr, y1_f, (int)T, (int)C, (int)C, (int)C, (int)C, (int)C, T * C, C * C, T * C);

  cvt_f32_bf16<<<dim3((unsigned)((M * C + 1023) / 1024)), b256, 0, stream>>>(y1_f, y1b, M * C, C, C);

  // out = y1 @ Wo^T + bo
  gemm_bf16<true><<<dim3((unsigned)(C / 64), (unsigned)(M / 64), 1), dim3(128), 0, stream>>>(
      y1b, wob, bo, out, (int)M, (int)C, (int)C, (int)C, (int)C, (int)C, 0, 0, 0);
}
